// AdvGNN_8160437862402
// MI455X (gfx1250) — compile-verified, matrix operand order checked by reading, not run on hardware
//
#include <hip/hip_runtime.h>

// ---------------------------------------------------------------------------
// GraphSAGE 2-layer inference for MI455X (gfx1250, wave32, WMMA).
//   h = relu(bn(mean_agg(x) @ Wl + b + x @ Wr))  (x2 layers)
// GEMMs run on v_wmma_f32_16x16x32_bf16 (bf16 in, fp32 accumulate).
// Scatter/degree are fp32 global atomics (bandwidth bound, ~23.3 TB/s roof).
// ---------------------------------------------------------------------------

typedef __bf16 bf16;
typedef bf16  v16bf __attribute__((ext_vector_type(16)));
typedef bf16  v8bf  __attribute__((ext_vector_type(8)));
typedef float v8f   __attribute__((ext_vector_type(8)));

#define HID   128
#define FPAD  160   // K padded to 5 chunks of 32 (covers F=130 and F=128)
#define LDT   168   // LDS row stride in bf16 (16B aligned, padded vs. 160)

static __device__ __forceinline__ v16bf cat8(v8bf lo, v8bf hi) {
  return __builtin_shufflevector(lo, hi, 0, 1, 2, 3, 4, 5, 6, 7,
                                         8, 9, 10, 11, 12, 13, 14, 15);
}

// ---------------------------------------------------------------------- zero
__global__ void zero_kernel(float* __restrict__ p, long long n) {
  long long i = (long long)blockIdx.x * blockDim.x + threadIdx.x;
  long long stride = (long long)gridDim.x * blockDim.x;
  for (; i < n; i += stride) p[i] = 0.0f;
}

// -------------------------------------------------- weights: fp32 (F,128) ->
// bf16 transposed [col][K] with zero pad to FPAD, so B-tiles load as b128.
__global__ void prep_weights(const float* __restrict__ W1l,
                             const float* __restrict__ W1r,
                             const float* __restrict__ W2l,
                             const float* __restrict__ W2r,
                             bf16* __restrict__ Wt, int F1, int F2) {
  int bi = blockIdx.x;           // 0..511
  int m  = bi >> 7;              // which matrix
  int n  = bi & 127;             // output column
  const float* W; int F;
  switch (m) {
    case 0:  W = W1l; F = F1; break;
    case 1:  W = W1r; F = F1; break;
    case 2:  W = W2l; F = F2; break;
    default: W = W2r; F = F2; break;
  }
  bf16* dst = Wt + (long long)m * HID * FPAD + (long long)n * FPAD;
  for (int k = threadIdx.x; k < FPAD; k += blockDim.x)
    dst[k] = (k < F) ? (bf16)W[(long long)k * HID + n] : (bf16)0.0f;
}

// ------------------------------------- fold bias + BN(eval) into alpha/beta:
// out = relu(acc*alpha + beta),  alpha=g*rsqrt(rv+eps), beta=(b-rm)*alpha+be
__global__ void prep_affine(const float* __restrict__ b1, const float* __restrict__ g1,
                            const float* __restrict__ be1, const float* __restrict__ rm1,
                            const float* __restrict__ rv1,
                            const float* __restrict__ b2, const float* __restrict__ g2,
                            const float* __restrict__ be2, const float* __restrict__ rm2,
                            const float* __restrict__ rv2,
                            float* __restrict__ alpha1, float* __restrict__ beta1,
                            float* __restrict__ alpha2, float* __restrict__ beta2) {
  int i = threadIdx.x;
  if (i < HID) {
    float a = g1[i] * rsqrtf(rv1[i] + 1e-5f);
    alpha1[i] = a;
    beta1[i]  = (b1[i] - rm1[i]) * a + be1[i];
  } else {
    int j = i - HID;
    float a = g2[j] * rsqrtf(rv2[j] + 1e-5f);
    alpha2[j] = a;
    beta2[j]  = (b2[j] - rm2[j]) * a + be2[j];
  }
}

// ------------------------------------------------------------------- degrees
__global__ void degree_kernel(const int* __restrict__ dstI,
                              float* __restrict__ deg, int E) {
  int e = blockIdx.x * blockDim.x + threadIdx.x;
  if (e < E) atomicAdd(&deg[dstI[e]], 1.0f);
}

// --------------------------------------------- message scatter (segment sum)
__global__ void scatter_kernel(const float* __restrict__ feat,
                               const int* __restrict__ srcI,
                               const int* __restrict__ dstI,
                               float* __restrict__ agg, int E, int F) {
  int e = blockIdx.x;
  if (e >= E) return;
  int s = srcI[e], d = dstI[e];
  const float* fp = feat + (long long)s * F;
  float*       ap = agg  + (long long)d * F;
  for (int f = threadIdx.x; f < F; f += blockDim.x)
    atomicAdd(&ap[f], fp[f]);
}

// -------------------------------------------------------------- fused GEMM:
// out[row, :] = relu( (agg[row]/max(deg,1)) @ Wl + x[row] @ Wr  folded-BN )
// 256 threads = 8 waves; block covers 16 rows x 128 cols; each wave owns a
// 16x16 tile, K-loop over 32-wide chunks with two bf16 WMMAs per chunk.
__global__ __launch_bounds__(256)
void sage_gemm_bn_relu(const float* __restrict__ agg,
                       const float* __restrict__ deg,
                       const float* __restrict__ xroot,
                       const bf16* __restrict__ WtL,   // [HID][FPAD] bf16
                       const bf16* __restrict__ WtR,   // [HID][FPAD] bf16
                       const float* __restrict__ alpha,
                       const float* __restrict__ beta,
                       float* __restrict__ out,
                       int N, int F) {
  __shared__ bf16 smM[16 * LDT];   // mean tile (zero padded to FPAD)
  __shared__ bf16 smX[16 * LDT];   // root tile

  const int tid     = threadIdx.x;
  const int rowBase = blockIdx.x * 16;

  // ---- cooperative stage: fp32 -> bf16, mean = agg * 1/max(deg,1) ----
  for (int i = tid; i < 16 * FPAD; i += 256) {
    int rr  = i / FPAD;
    int kk  = i - rr * FPAD;
    int row = rowBase + rr;
    float m = 0.0f, xv = 0.0f;
    if (row < N && kk < F) {
      float rinv = 1.0f / fmaxf(deg[row], 1.0f);
      m  = agg[(long long)row * F + kk] * rinv;
      xv = xroot[(long long)row * F + kk];
    }
    smM[rr * LDT + kk] = (bf16)m;
    smX[rr * LDT + kk] = (bf16)xv;
  }
  __syncthreads();

  // ---- per-wave 16x16 WMMA tile ----
  const int lane    = tid & 31;
  const int wv      = tid >> 5;        // 0..7 -> column tile
  const int half    = lane >> 4;       // 0: K 0..7/16..23, 1: K 8..15/24..31
  const int r       = lane & 15;       // A row / B column within tile
  const int colBase = wv * 16;
  const int col     = colBase + r;

  v8f c = {};
  const int chunks = (F + 31) >> 5;    // 5 for F=130, 4 for F=128
  const bf16* bl = WtL + (long long)col * FPAD;
  const bf16* br = WtR + (long long)col * FPAD;

  for (int ch = 0; ch < chunks; ++ch) {
    const int kA = ch * 32 + half * 8;

    // mean @ Wl
    {
      v16bf a = cat8(*(const v8bf*)&smM[r * LDT + kA],
                     *(const v8bf*)&smM[r * LDT + kA + 16]);
      v16bf b = cat8(*(const v8bf*)&bl[kA],
                     *(const v8bf*)&bl[kA + 16]);
      c = __builtin_amdgcn_wmma_f32_16x16x32_bf16(false, a, false, b,
                                                  (short)0, c, false, false);
    }
    // x @ Wr
    {
      v16bf a = cat8(*(const v8bf*)&smX[r * LDT + kA],
                     *(const v8bf*)&smX[r * LDT + kA + 16]);
      v16bf b = cat8(*(const v8bf*)&br[kA],
                     *(const v8bf*)&br[kA + 16]);
      c = __builtin_amdgcn_wmma_f32_16x16x32_bf16(false, a, false, b,
                                                  (short)0, c, false, false);
    }
  }

  // ---- epilogue: folded bias+BN then ReLU; D layout: vgpr i -> M=i(+8) ----
  const float al = alpha[col];
  const float be = beta[col];
#pragma unroll
  for (int i = 0; i < 8; ++i) {
    int row = rowBase + i + half * 8;
    if (row < N) {
      float v = c[i] * al + be;
      out[(long long)row * HID + col] = fmaxf(v, 0.0f);
    }
  }
}

// ---------------------------------------------------------------------------
extern "C" void kernel_launch(void* const* d_in, const int* in_sizes, int n_in,
                              void* d_out, int out_size, void* d_ws, size_t ws_size,
                              hipStream_t stream) {
  const float* x   = (const float*)d_in[0];
  const int*   ei  = (const int*)d_in[1];
  const float* W1l = (const float*)d_in[2];
  const float* b1l = (const float*)d_in[3];
  const float* W1r = (const float*)d_in[4];
  const float* g1  = (const float*)d_in[5];
  const float* be1 = (const float*)d_in[6];
  const float* rm1 = (const float*)d_in[7];
  const float* rv1 = (const float*)d_in[8];
  const float* W2l = (const float*)d_in[9];
  const float* b2l = (const float*)d_in[10];
  const float* W2r = (const float*)d_in[11];
  const float* g2  = (const float*)d_in[12];
  const float* be2 = (const float*)d_in[13];
  const float* rm2 = (const float*)d_in[14];
  const float* rv2 = (const float*)d_in[15];

  const int IND = 130;
  const int N = in_sizes[0] / IND;
  const int E = in_sizes[1] / 2;
  const int* srcI = ei;
  const int* dstI = ei + E;

  // ---- carve workspace ----
  char*  ws  = (char*)d_ws;
  size_t off = 0;
  auto alloc = [&](size_t bytes) -> void* {
    void* p = ws + off;
    off = (off + bytes + 255) & ~(size_t)255;
    return p;
  };
  float* deg = (float*)alloc((size_t)N * sizeof(float));
  float* agg = (float*)alloc((size_t)N * IND * sizeof(float));   // reused L2
  float* h1  = (float*)alloc((size_t)N * HID * sizeof(float));
  bf16*  Wt  = (bf16*) alloc((size_t)4 * HID * FPAD * sizeof(bf16));
  float* aff = (float*)alloc((size_t)4 * HID * sizeof(float));
  float* alpha1 = aff;
  float* beta1  = aff + HID;
  float* alpha2 = aff + 2 * HID;
  float* beta2  = aff + 3 * HID;

  // ---- prep ----
  zero_kernel<<<1024, 256, 0, stream>>>(deg, (long long)N);
  zero_kernel<<<4096, 256, 0, stream>>>(agg, (long long)N * IND);
  prep_weights<<<512, FPAD, 0, stream>>>(W1l, W1r, W2l, W2r, Wt, IND, HID);
  prep_affine<<<1, 256, 0, stream>>>(b1l, g1, be1, rm1, rv1,
                                     b2l, g2, be2, rm2, rv2,
                                     alpha1, beta1, alpha2, beta2);
  degree_kernel<<<(E + 255) / 256, 256, 0, stream>>>(dstI, deg, E);

  // ---- layer 1 ----
  scatter_kernel<<<E, 128, 0, stream>>>(x, srcI, dstI, agg, E, IND);
  sage_gemm_bn_relu<<<(N + 15) / 16, 256, 0, stream>>>(
      agg, deg, x, Wt, Wt + (size_t)1 * HID * FPAD, alpha1, beta1, h1, N, IND);

  // ---- layer 2 ----
  zero_kernel<<<4096, 256, 0, stream>>>(agg, (long long)N * HID);
  scatter_kernel<<<E, 128, 0, stream>>>(h1, srcI, dstI, agg, E, HID);
  sage_gemm_bn_relu<<<(N + 15) / 16, 256, 0, stream>>>(
      agg, deg, h1, Wt + (size_t)2 * HID * FPAD, Wt + (size_t)3 * HID * FPAD,
      alpha2, beta2, (float*)d_out, N, HID);
}
